// AutoregressiveFormulaDecoder_73306501808222
// MI455X (gfx1250) — matrix operand order repaired and checked by reading, not compiled, not run on hardware
//
#include <hip/hip_runtime.h>

typedef __bf16 bf16_t;
typedef __attribute__((ext_vector_type(16))) __bf16 v16bf;
typedef __attribute__((ext_vector_type(8)))  float  v8f;

#define BATCH   8192
#define HID     256
#define NSTEPS  49
#define TOKT    50
#define VOCAB   148
#define VOCABP  160
#define LATENT  32
#define START_TOK 1
#define BT      64              // batch rows per workgroup (4 waves x 16 rows)
#define LROW    (HID + 8)       // padded LDS row stride in bf16 elements
#define BUFSZ   ((unsigned)(BT * LROW))   // one h buffer, in bf16 elements

// LDS buffer offsets (bf16 element units) off a single base pointer.
// Ping-pong is done with integer offsets (NOT pointer selects) so that
// InferAddressSpaces keeps every access in addrspace(3) -> pure ds_* ops.
#define OFF_A   (0u)            // h0 ping
#define OFF_B   (1u * BUFSZ)    // h0 pong
#define OFF_C   (2u * BUFSZ)    // h1 ping
#define OFF_D   (3u * BUFSZ)    // h1 pong

// ---------------------------------------------------------------------------
// fragment loaders
// A-matrix 16x32 bf16 (ISA 7.12.2): lane m=lane&15, half=lane>>4;
//   regs 0..3 hold K = kbase+half*8 .. +7 (pairs), regs 4..7 hold K = kbase+16+half*8 .. +7
//   -> contiguous 24-element window, take [0..7] and [16..23]
// B-matrix 32x16 bf16: lane n=lane&15; same K order as A: [kbase+half*8 .. +7] and [kbase+16+half*8 .. +7]
// ---------------------------------------------------------------------------
union FragU { uint4 u[2]; v16bf v; };

__device__ __forceinline__ v16bf load_a_glb(const bf16_t* prow, int half, int kbase) {
  const bf16_t* p = prow + kbase + half * 8;
  FragU f;
  f.u[0] = *(const uint4*)(p);
  f.u[1] = *(const uint4*)(p + 16);
  return f.v;
}
__device__ __forceinline__ v16bf load_a_lds(const bf16_t* lds, unsigned idx) {
  FragU f;
  f.u[0] = *(const uint4*)(lds + idx);
  f.u[1] = *(const uint4*)(lds + idx + 16);
  return f.v;
}
__device__ __forceinline__ v16bf load_b_frag(const bf16_t* W, int rowlen, int n, int kbase, int half) {
  const bf16_t* p = W + (size_t)n * rowlen + kbase + half * 8;
  FragU f;
  f.u[0] = *(const uint4*)(p);
  f.u[1] = *(const uint4*)(p + 16);
  return f.v;
}

__device__ __forceinline__ v8f wmma_bf16(v16bf a, v16bf b, v8f c) {
  return __builtin_amdgcn_wmma_f32_16x16x32_bf16(false, a, false, b, (short)0, c, false, false);
}

__device__ __forceinline__ float sigm_f(float x) { return 1.f / (1.f + __expf(-x)); }
__device__ __forceinline__ float tanh_f(float x) { return 2.f / (1.f + __expf(-2.f * x)) - 1.f; }

// ---------------------------------------------------------------------------
// One GRU layer update for a wave's 16 batch rows, all 256 hidden columns.
// X_LDS=false: x rows come from global (embedding row per lane, xrow_g)
// X_LDS=true : x rows come from LDS at element offset xoff (per-lane row base)
// hs/hd: LDS element offsets of h source / destination buffers (WG base)
// ---------------------------------------------------------------------------
template<bool X_LDS>
__device__ __forceinline__ void gru_layer(
    bf16_t* lds, const bf16_t* __restrict__ xrow_g, unsigned xoff,
    unsigned hs, unsigned hd,
    const bf16_t* __restrict__ Wih, const bf16_t* __restrict__ Whh,
    const float* __restrict__ bih, const float* __restrict__ bhh,
    int wbase, int lm, int half)
{
  const unsigned hrow = hs + (unsigned)(wbase + lm) * LROW + (unsigned)(half * 8);
  const unsigned xrow_l = xoff + (unsigned)(half * 8);
#pragma unroll 1
  for (int nt = 0; nt < HID / 16; ++nt) {
    int col = nt * 16 + lm;
    v8f aR = {}, aZ = {}, aNi = {}, aNh = {};
#pragma unroll
    for (int kt = 0; kt < HID / 32; ++kt) {
      int kb = kt * 32;
      v16bf ax = X_LDS ? load_a_lds(lds, xrow_l + kb)
                       : load_a_glb(xrow_g, half, kb);
      v16bf ah = load_a_lds(lds, hrow + kb);
      aR  = wmma_bf16(ax, load_b_frag(Wih, HID, col,           kb, half), aR);
      aZ  = wmma_bf16(ax, load_b_frag(Wih, HID, HID + col,     kb, half), aZ);
      aNi = wmma_bf16(ax, load_b_frag(Wih, HID, 2 * HID + col, kb, half), aNi);
      aR  = wmma_bf16(ah, load_b_frag(Whh, HID, col,           kb, half), aR);
      aZ  = wmma_bf16(ah, load_b_frag(Whh, HID, HID + col,     kb, half), aZ);
      aNh = wmma_bf16(ah, load_b_frag(Whh, HID, 2 * HID + col, kb, half), aNh);
    }
    float br  = bih[col] + bhh[col];
    float bz  = bih[HID + col] + bhh[HID + col];
    float bni = bih[2 * HID + col];
    float bnh = bhh[2 * HID + col];
#pragma unroll
    for (int r = 0; r < 8; ++r) {
      int m = r + 8 * half;                    // C/D layout: row = reg + 8*half
      float rg = sigm_f(aR[r] + br);
      float zg = sigm_f(aZ[r] + bz);
      float ng = tanh_f(aNi[r] + bni + rg * (aNh[r] + bnh));
      unsigned cell = (unsigned)(wbase + m) * LROW + (unsigned)col;
      float hold = (float)lds[hs + cell];
      float hnew = (1.f - zg) * ng + zg * hold;
      lds[hd + cell] = (bf16_t)hnew;
    }
  }
}

// ---------------------------------------------------------------------------
// Persistent decoder kernel: each wave owns 16 batch rows for all 49 steps.
// No barriers: a wave only ever touches its own 16 LDS rows.
// ---------------------------------------------------------------------------
__global__ void __launch_bounds__(128) gru_decoder_kernel(
    const float* __restrict__ z,
    const int*   __restrict__ target,
    const float* __restrict__ b_lat,
    const float* __restrict__ b_ih0, const float* __restrict__ b_hh0,
    const float* __restrict__ b_ih1, const float* __restrict__ b_hh1,
    const float* __restrict__ b1,    const float* __restrict__ b2,
    const bf16_t* __restrict__ embB,  const bf16_t* __restrict__ WlatB,
    const bf16_t* __restrict__ Wih0B, const bf16_t* __restrict__ Whh0B,
    const bf16_t* __restrict__ Wih1B, const bf16_t* __restrict__ Whh1B,
    const bf16_t* __restrict__ W1B,   const bf16_t* __restrict__ W2P,
    float* __restrict__ logits)
{
  extern __shared__ bf16_t lds[];

  const int lane  = threadIdx.x & 31;
  const int wave  = threadIdx.x >> 5;
  const int half  = lane >> 4;
  const int lm    = lane & 15;
  const int wbase = wave * 16;
  const int blk0  = blockIdx.x * BT;
  const int browA = blk0 + wbase + lm;          // batch row this lane feeds as A row
  const unsigned lrow = (unsigned)(wbase + lm) * LROW;   // per-lane A-row base (elements)

  // ---- initial hidden state: h = z @ W_lat^T + b_lat  (K = 32, one WMMA/tile)
  v16bf az;
  {
    const float* zp = z + (size_t)browA * LATENT + half * 8;
#pragma unroll
    for (int j = 0; j < 8; ++j) {
      az[j]     = (bf16_t)zp[j];
      az[8 + j] = (bf16_t)zp[16 + j];
    }
  }
#pragma unroll 1
  for (int nt = 0; nt < (2 * HID) / 16; ++nt) {
    int o = nt * 16 + lm;                       // output col in [0, 512)
    v8f acc = {};
    acc = wmma_bf16(az, load_b_frag(WlatB, LATENT, o, 0, half), acc);
    float bb = b_lat[o];
    unsigned dsto = (o < HID) ? OFF_A : OFF_C;  // integer offset select -> stays ds_*
    unsigned c = (unsigned)(o & (HID - 1));
#pragma unroll
    for (int r = 0; r < 8; ++r) {
      int m = r + 8 * half;
      lds[dsto + (unsigned)(wbase + m) * LROW + c] = (bf16_t)(acc[r] + bb);
    }
  }

#pragma unroll 1
  for (int t = 0; t < NSTEPS; ++t) {
    // ping-pong via integer offsets only
    unsigned h0s = (t & 1) ? OFF_B : OFF_A;
    unsigned h0d = (t & 1) ? OFF_A : OFF_B;
    unsigned h1s = (t & 1) ? OFF_D : OFF_C;
    unsigned h1d = (t & 1) ? OFF_C : OFF_D;
    unsigned hdo = h0s;                         // h0 source buffer is dead after layer 0

    // tokens_in[:, 0] = START ; tokens_in[:, j] = target[:, j] for j >= 1
    int tok = (t == 0) ? START_TOK : target[(size_t)browA * TOKT + t];
    const bf16_t* xrow = embB + (size_t)tok * HID;

    gru_layer<false>(lds, xrow, 0u, h0s, h0d, Wih0B, Whh0B, b_ih0, b_hh0, wbase, lm, half);
    gru_layer<true >(lds, nullptr, h0d + lrow, h1s, h1d, Wih1B, Whh1B, b_ih1, b_hh1, wbase, lm, half);

    // hdd = relu(h1' @ W1^T + b1)
    const unsigned h1drow = h1d + lrow + (unsigned)(half * 8);
#pragma unroll 1
    for (int nt = 0; nt < HID / 16; ++nt) {
      int col = nt * 16 + lm;
      v8f acc = {};
#pragma unroll
      for (int kt = 0; kt < HID / 32; ++kt) {
        int kb = kt * 32;
        v16bf a = load_a_lds(lds, h1drow + kb);
        acc = wmma_bf16(a, load_b_frag(W1B, HID, col, kb, half), acc);
      }
      float bb = b1[col];
#pragma unroll
      for (int r = 0; r < 8; ++r) {
        int m = r + 8 * half;
        float v = acc[r] + bb;
        lds[hdo + (unsigned)(wbase + m) * LROW + (unsigned)col] =
            (bf16_t)(v > 0.f ? v : 0.f);
      }
    }

    // logits = hdd @ W2^T + b2   (W2 padded to 160 rows of zeros)
    const unsigned hdrow = hdo + lrow + (unsigned)(half * 8);
#pragma unroll 1
    for (int vt = 0; vt < VOCABP / 16; ++vt) {
      int v = vt * 16 + lm;
      v8f acc = {};
#pragma unroll
      for (int kt = 0; kt < HID / 32; ++kt) {
        int kb = kt * 32;
        v16bf a = load_a_lds(lds, hdrow + kb);
        acc = wmma_bf16(a, load_b_frag(W2P, HID, v, kb, half), acc);
      }
      if (v < VOCAB) {
        float bb = b2[v];
#pragma unroll
        for (int r = 0; r < 8; ++r) {
          int m = r + 8 * half;
          size_t b = (size_t)(blk0 + wbase + m);
          logits[(b * NSTEPS + t) * VOCAB + v] = acc[r] + bb;
        }
      }
    }
  }
}

// ---------------------------------------------------------------------------
// prep / epilogue kernels
// ---------------------------------------------------------------------------
__global__ void cvt_f32_bf16(const float* __restrict__ s, bf16_t* __restrict__ d, int n) {
  int i = blockIdx.x * 256 + threadIdx.x;
  if (i < n) d[i] = (bf16_t)s[i];
}
__global__ void cvt_pad_w2(const float* __restrict__ w2, bf16_t* __restrict__ d) {
  int i = blockIdx.x * 256 + threadIdx.x;
  if (i < VOCABP * HID) {
    int row = i / HID;
    d[i] = (row < VOCAB) ? (bf16_t)w2[i] : (bf16_t)0.f;
  }
}
__global__ void gen_tokens(const int* __restrict__ target, float* __restrict__ out) {
  int i = blockIdx.x * 256 + threadIdx.x;
  if (i < BATCH * NSTEPS) {
    int b = i / NSTEPS, j = i % NSTEPS;
    out[i] = (float)target[(size_t)b * TOKT + j + 1];
  }
}

// ---------------------------------------------------------------------------
extern "C" void kernel_launch(void* const* d_in, const int* in_sizes, int n_in,
                              void* d_out, int out_size, void* d_ws, size_t ws_size,
                              hipStream_t stream) {
  (void)in_sizes; (void)n_in; (void)out_size; (void)ws_size;
  const float* z      = (const float*)d_in[0];
  const int*   target = (const int*)d_in[1];
  const float* emb    = (const float*)d_in[2];
  const float* W_lat  = (const float*)d_in[3];
  const float* b_lat  = (const float*)d_in[4];
  const float* W_ih0  = (const float*)d_in[5];
  const float* W_hh0  = (const float*)d_in[6];
  const float* b_ih0  = (const float*)d_in[7];
  const float* b_hh0  = (const float*)d_in[8];
  const float* W_ih1  = (const float*)d_in[9];
  const float* W_hh1  = (const float*)d_in[10];
  const float* b_ih1  = (const float*)d_in[11];
  const float* b_hh1  = (const float*)d_in[12];
  const float* W1     = (const float*)d_in[13];
  const float* b1     = (const float*)d_in[14];
  const float* W2     = (const float*)d_in[15];
  const float* b2     = (const float*)d_in[16];

  // workspace layout: bf16 copies of weights (all 16B-aligned)
  bf16_t* p = (bf16_t*)d_ws;
  bf16_t* embB  = p; p += VOCAB * HID;
  bf16_t* WlatB = p; p += 2 * HID * LATENT;
  bf16_t* Wih0B = p; p += 3 * HID * HID;
  bf16_t* Whh0B = p; p += 3 * HID * HID;
  bf16_t* Wih1B = p; p += 3 * HID * HID;
  bf16_t* Whh1B = p; p += 3 * HID * HID;
  bf16_t* W1B   = p; p += HID * HID;
  bf16_t* W2P   = p; p += VOCABP * HID;

  auto cvt = [&](const float* s, bf16_t* d, int n) {
    cvt_f32_bf16<<<(n + 255) / 256, 256, 0, stream>>>(s, d, n);
  };
  cvt(emb,   embB,  VOCAB * HID);
  cvt(W_lat, WlatB, 2 * HID * LATENT);
  cvt(W_ih0, Wih0B, 3 * HID * HID);
  cvt(W_hh0, Whh0B, 3 * HID * HID);
  cvt(W_ih1, Wih1B, 3 * HID * HID);
  cvt(W_hh1, Whh1B, 3 * HID * HID);
  cvt(W1,    W1B,   HID * HID);
  cvt_pad_w2<<<(VOCABP * HID + 255) / 256, 256, 0, stream>>>(W2, W2P);

  float* logits = (float*)d_out;
  float* gen    = logits + (size_t)BATCH * NSTEPS * VOCAB;
  gen_tokens<<<(BATCH * NSTEPS + 255) / 256, 256, 0, stream>>>(target, gen);

  size_t smem = 4ull * BT * LROW * sizeof(bf16_t);   // ~132 KB -> 2 WGs / WGP
  gru_decoder_kernel<<<BATCH / BT, 128, smem, stream>>>(
      z, target, b_lat, b_ih0, b_hh0, b_ih1, b_hh1, b1, b2,
      embB, WlatB, Wih0B, Whh0B, Wih1B, Whh1B, W1B, W2P, logits);
}
